// PerformerAttention_49838800503120
// MI455X (gfx1250) — matrix operand order repaired and checked by reading, not compiled, not run on hardware
//
#include <hip/hip_runtime.h>
#include <hip/hip_bf16.h>

typedef __attribute__((ext_vector_type(16))) _Float16 v16h;
typedef __attribute__((ext_vector_type(8)))  float    v8f;

#define BB 4
#define HH 12
#define BH 48
#define NN 4096
#define DD 64
#define MM 256
#define EPSF   1e-4f
#define RATIO  0.0625f   /* M^-0.5 = 1/16  */
#define XSCALE 0.125f    /* D^-0.5 = 1/8 (= s*dn) */

// ---------- ordered-float encoding for atomicMax on mixed-sign floats ----------
__device__ __forceinline__ unsigned fenc(float f) {
    unsigned u = __float_as_uint(f);
    return (u & 0x80000000u) ? ~u : (u | 0x80000000u);
}
__device__ __forceinline__ float fdec(unsigned e) {
    unsigned u = (e & 0x80000000u) ? (e & 0x7FFFFFFFu) : ~e;
    return __uint_as_float(u);
}

// ---------- WMMA fragment loaders (wave32; A/B of each product use the same k order) ----------
// A (16x32 f16): lane = 16*hi + row; halfs h -> K = k0+16*hi + h (contiguous 16)
__device__ __forceinline__ v16h load_a_x(const float* base, int row0, int k0,
                                         const float* mrow /*may be null*/) {
    int lane = threadIdx.x & 31;
    int r = lane & 15, hi = lane >> 4;
    float sc = XSCALE;
    if (mrow) sc *= mrow[row0 + r];
    const float* p = base + (size_t)(row0 + r) * DD + k0 + 16 * hi;
    v16h a;
#pragma unroll
    for (int j = 0; j < 8; ++j) a[j]     = (_Float16)(p[j]      * sc);
#pragma unroll
    for (int j = 0; j < 8; ++j) a[8 + j] = (_Float16)(p[8 + j] * sc);
    return a;
}

// B (32x16 f16): lane = 16*hi + col; halfs h -> K = 16*hi + h (contiguous 16)
// B = proj^T : B[k=d][col=m], proj stored (M,D) row-major -> contiguous in d.
__device__ __forceinline__ v16h load_b_projT(const float* proj, int mtile, int kc) {
    int lane = threadIdx.x & 31;
    int m = mtile * 16 + (lane & 15);
    int hi = lane >> 4;
    const float* p = proj + (size_t)m * DD + kc + 16 * hi;
    v16h b;
#pragma unroll
    for (int j = 0; j < 16; ++j) b[j] = (_Float16)p[j];
    return b;
}

// B = V tile : B[k=n][col=d], V stored (N,D) row-major -> strided in n.
__device__ __forceinline__ v16h load_b_v(const float* Vbh, int n0, int dtile,
                                         const float* mrow) {
    int lane = threadIdx.x & 31;
    int d = dtile * 16 + (lane & 15);
    int hi = lane >> 4;
    const float* p = Vbh + (size_t)(n0 + 8 * hi) * DD + d;
    v16h b;
#pragma unroll
    for (int j = 0; j < 16; ++j) {
        int kk = (j & 7) + ((j >> 3) << 4);
        float mv = mrow ? mrow[n0 + 8 * hi + kk] : 1.0f;
        b[j] = (_Float16)(p[(size_t)kk * DD] * mv);
    }
    return b;
}

// one 16x16 tile of data_dash = (X * XSCALE [*mask]) @ proj^T  (K = D = 64)
__device__ __forceinline__ v8f qk_dash_tile(const float* Xbh, const float* proj,
                                            int n0, int mtile, const float* mrow) {
    v8f acc = {};
    v16h a0 = load_a_x(Xbh, n0, 0, mrow);
    v16h b0 = load_b_projT(proj, mtile, 0);
    acc = __builtin_amdgcn_wmma_f32_16x16x32_f16(false, a0, false, b0,
                                                 (short)0, acc, false, false);
    v16h a1 = load_a_x(Xbh, n0, 32, mrow);
    v16h b1 = load_b_projT(proj, mtile, 32);
    acc = __builtin_amdgcn_wmma_f32_16x16x32_f16(false, a1, false, b1,
                                                 (short)0, acc, false, false);
    return acc;
}

// ---------------- kernel 0: init accumulators in ws ----------------
__global__ void perf_init(unsigned* gmax, float* zacc, int n) {
    int i = blockIdx.x * 256 + threadIdx.x;
    if (i == 0) *gmax = 0u;                 // fenc of very negative
    if (i < n) zacc[i] = 0.0f;
}

// ---------------- kernel 1: global max of K's data_dash ----------------
__global__ void perf_kmax(const float* K, const float* proj, const float* mask,
                          unsigned* gmax) {
    int blk = blockIdx.x;
    int bh = blk >> 8, nt = blk & 255;
    const float* Kbh = K + (size_t)bh * NN * DD;
    const float* mrow = mask + (size_t)(bh / HH) * NN;
    int wave = threadIdx.x >> 5;
    int n0 = nt * 16;
    __shared__ unsigned smax;
    if (threadIdx.x == 0) smax = 0u;
    __syncthreads();
    float mx = -3.402823466e+38f;
#pragma unroll
    for (int t = 0; t < 2; ++t) {
        v8f acc = qk_dash_tile(Kbh, proj, n0, wave * 2 + t, mrow);
#pragma unroll
        for (int r = 0; r < 8; ++r) mx = fmaxf(mx, acc[r]);
    }
#pragma unroll
    for (int off = 1; off < 32; off <<= 1) mx = fmaxf(mx, __shfl_xor(mx, off, 32));
    if ((threadIdx.x & 31) == 0) atomicMax(&smax, fenc(mx));
    __syncthreads();
    if (threadIdx.x == 0) atomicMax(gmax, smax);
}

// ---------------- kernel 2: k_prime -> k_sum, context ----------------
// grid (BH, 16 splits), 256 threads; each block handles 256 rows as 8 x 32.
__global__ void perf_ctx(const float* K, const float* V, const float* proj,
                         const float* mask, const unsigned* gmax_enc,
                         float* ksum, float* ctx) {
    int bh = blockIdx.x, split = blockIdx.y;
    const float* Kbh = K + (size_t)bh * NN * DD;
    const float* Vbh = V + (size_t)bh * NN * DD;
    const float* mrow = mask + (size_t)(bh / HH) * NN;
    float* ksum_bh = ksum + (size_t)bh * MM;
    float* ctx_bh  = ctx  + (size_t)bh * MM * DD;
    float gmax = fdec(*gmax_enc);
    int lane = threadIdx.x & 31, wave = threadIdx.x >> 5, hi = lane >> 4;

    __shared__ float diag[32];
    v8f cacc[2][4];
#pragma unroll
    for (int a = 0; a < 2; ++a)
#pragma unroll
        for (int b = 0; b < 4; ++b) cacc[a][b] = (v8f){};
    float ks[2] = {0.0f, 0.0f};

    int nbase = split * 256;
    for (int it = 0; it < 8; ++it) {
        int n0 = nbase + it * 32;
        __syncthreads();
        if (threadIdx.x < 32) diag[threadIdx.x] = 0.0f;
        __syncthreads();
        {   // diag[n] = 0.5 * sum_d x^2 over 32 rows, 8 d each
            int row = threadIdx.x & 31, ds = (threadIdx.x >> 5) * 8;
            float mk = XSCALE * mrow[n0 + row];
            const float* p = Kbh + (size_t)(n0 + row) * DD + ds;
            float s = 0.0f;
#pragma unroll
            for (int j = 0; j < 8; ++j) { float v = p[j] * mk; s += v * v; }
            atomicAdd(&diag[row], 0.5f * s);
        }
        __syncthreads();

        v16h bv[4];
#pragma unroll
        for (int dt = 0; dt < 4; ++dt) bv[dt] = load_b_v(Vbh, n0, dt, mrow);

#pragma unroll
        for (int mt = 0; mt < 2; ++mt) {
            int mtile = wave * 2 + mt;
            v8f d0 = qk_dash_tile(Kbh, proj, n0,      mtile, mrow);
            v8f d1 = qk_dash_tile(Kbh, proj, n0 + 16, mtile, mrow);
            // D-layout of two n-tiles == f16 A-layout of k_prime^T (K=32)
            v16h af; float ls = 0.0f;
#pragma unroll
            for (int r = 0; r < 8; ++r) {
                float e0 = RATIO * (__expf(d0[r] - diag[r + 8 * hi]      - gmax) + EPSF);
                float e1 = RATIO * (__expf(d1[r] - diag[16 + r + 8 * hi] - gmax) + EPSF);
                ls += e0 + e1;
                af[r] = (_Float16)e0; af[8 + r] = (_Float16)e1;
            }
            ks[mt] += ls;
#pragma unroll
            for (int dt = 0; dt < 4; ++dt)
                cacc[mt][dt] = __builtin_amdgcn_wmma_f32_16x16x32_f16(
                    false, af, false, bv[dt], (short)0, cacc[mt][dt], false, false);
        }
    }
    // k_sum: lanes L and L+16 hold the same m
#pragma unroll
    for (int mt = 0; mt < 2; ++mt) {
        float s = ks[mt] + __shfl_xor(ks[mt], 16, 32);
        if (lane < 16) atomicAdd(&ksum_bh[(wave * 2 + mt) * 16 + lane], s);
    }
    // context: D rows = m, cols = d
#pragma unroll
    for (int mt = 0; mt < 2; ++mt)
#pragma unroll
        for (int dt = 0; dt < 4; ++dt)
#pragma unroll
            for (int r = 0; r < 8; ++r) {
                int m = (wave * 2 + mt) * 16 + r + 8 * hi;
                int d = dt * 16 + (lane & 15);
                atomicAdd(&ctx_bh[(size_t)m * DD + d], cacc[mt][dt][r]);
            }
}

// ---------------- kernel 3: context (M,D) f32 -> context^T (D,M) f16 --------
__global__ void perf_tctx(const float* ctx, _Float16* ctxT) {
    size_t i = (size_t)blockIdx.x * 256 + threadIdx.x;   // over BH*DD*MM
    int m = (int)(i % MM);
    size_t t = i / MM;
    int d = (int)(t % DD);
    size_t bh = t / DD;
    ctxT[i] = (_Float16)ctx[(bh * MM + m) * DD + d];
}

// ---------------- kernel 4: q_prime, denominator, out ----------------
__global__ void perf_out(const float* Q, const float* proj, const float* ksum,
                         const _Float16* ctxT, float* out) {
    int blk = blockIdx.x;
    int bh = blk >> 8, nt = blk & 255;
    int n0 = nt * 16;
    const float* Qbh = Q + (size_t)bh * NN * DD;
    const _Float16* ctxTbh = ctxT + (size_t)bh * DD * MM;
    const float* ksum_bh = ksum + (size_t)bh * MM;
    float* out_bh = out + (size_t)bh * NN * DD;
    int lane = threadIdx.x & 31, wave = threadIdx.x >> 5, hi = lane >> 4;

    __shared__ float     diag[16];
    __shared__ unsigned  rmax[16];
    __shared__ float     dsum[16];
    __shared__ _Float16  qp[16][MM];     // 8 KB
    __shared__ float     obuf[4 * 8 * 32];

    if (threadIdx.x < 16) { diag[threadIdx.x] = 0.0f; rmax[threadIdx.x] = 0u; dsum[threadIdx.x] = 0.0f; }
    __syncthreads();
    {   // diag over 16 rows, 4 d per thread
        int row = threadIdx.x & 15, ds = (threadIdx.x >> 4) * 4;
        const float* p = Qbh + (size_t)(n0 + row) * DD + ds;
        float s = 0.0f;
#pragma unroll
        for (int j = 0; j < 4; ++j) { float v = p[j] * XSCALE; s += v * v; }
        atomicAdd(&diag[row], 0.5f * s);
    }
    __syncthreads();

    // GEMM1: q_dash, 2 m-tiles per wave
    v8f qd[2];
#pragma unroll
    for (int mt = 0; mt < 2; ++mt)
        qd[mt] = qk_dash_tile(Qbh, proj, n0, wave * 2 + mt, nullptr);

    // rowwise max over m (butterfly in 16-lane groups, then shared atomicMax)
#pragma unroll
    for (int r = 0; r < 8; ++r) {
        float v = fmaxf(qd[0][r], qd[1][r]);
#pragma unroll
        for (int off = 1; off < 16; off <<= 1) v = fmaxf(v, __shfl_xor(v, off, 32));
        if ((lane & 15) == 0) atomicMax(&rmax[r + 8 * hi], fenc(v));
    }
    __syncthreads();

    // q_prime, denominator, stage q_prime (f16) to LDS
#pragma unroll
    for (int mt = 0; mt < 2; ++mt) {
        int mtile = wave * 2 + mt;
        float kv = ksum_bh[mtile * 16 + (lane & 15)];
#pragma unroll
        for (int r = 0; r < 8; ++r) {
            int n = r + 8 * hi;
            float v = RATIO * (__expf(qd[mt][r] - diag[n] - fdec(rmax[n])) + EPSF);
            qp[n][mtile * 16 + (lane & 15)] = (_Float16)v;
            float dv = v * kv;
#pragma unroll
            for (int off = 1; off < 16; off <<= 1) dv += __shfl_xor(dv, off, 32);
            if ((lane & 15) == 0) atomicAdd(&dsum[n], dv);
        }
    }
    __syncthreads();

    // GEMM2: out = q_prime @ context ; wave -> (d-tile, m-half)
    int dt = wave & 3, mh = wave >> 2;
    v8f oacc = {};
#pragma unroll
    for (int c = 0; c < 4; ++c) {
        int mc = mh * 128 + c * 32;
        v16h a;
        {   // A from LDS q_prime: lane row = lane&15, two 8-half runs
            int n = lane & 15;
            const _Float16* p = &qp[n][mc + 8 * hi];
#pragma unroll
            for (int j = 0; j < 8; ++j) a[j]     = p[j];
#pragma unroll
            for (int j = 0; j < 8; ++j) a[8 + j] = p[16 + j];
        }
        v16h b;
        {   // B from context^T (D,M) f16: two 8-half runs
            int d = dt * 16 + (lane & 15);
            const _Float16* p = ctxTbh + (size_t)d * MM + mc + 8 * hi;
#pragma unroll
            for (int j = 0; j < 16; ++j) b[j] = p[(j & 7) + ((j >> 3) << 4)];
        }
        oacc = __builtin_amdgcn_wmma_f32_16x16x32_f16(false, a, false, b,
                                                      (short)0, oacc, false, false);
    }
    if (mh == 0) {
#pragma unroll
        for (int r = 0; r < 8; ++r) obuf[dt * 256 + r * 32 + lane] = oacc[r];
    }
    __syncthreads();
    if (mh == 1) {
#pragma unroll
        for (int r = 0; r < 8; ++r) {
            int n = r + 8 * hi;
            float val = oacc[r] + obuf[dt * 256 + r * 32 + lane];
            val *= 1.0f / dsum[n];
            out_bh[(size_t)(n0 + n) * DD + dt * 16 + (lane & 15)] = val;
        }
    }
}

// ---------------- launch ----------------
extern "C" void kernel_launch(void* const* d_in, const int* in_sizes, int n_in,
                              void* d_out, int out_size, void* d_ws, size_t ws_size,
                              hipStream_t stream) {
    const float* Q    = (const float*)d_in[0];
    const float* K    = (const float*)d_in[1];
    const float* V    = (const float*)d_in[2];
    const float* mask = (const float*)d_in[3];
    const float* proj = (const float*)d_in[4];
    float* out = (float*)d_out;

    unsigned* gmax = (unsigned*)d_ws;
    float* ksum = (float*)d_ws + 16;
    float* ctx  = ksum + (size_t)BH * MM;
    _Float16* ctxT = (_Float16*)(ctx + (size_t)BH * MM * DD);

    int zn = BH * MM + BH * MM * DD;
    perf_init<<<(zn + 255) / 256, 256, 0, stream>>>(gmax, ksum, zn);
    perf_kmax<<<BH * (NN / 16), 256, 0, stream>>>(K, proj, mask, gmax);
    perf_ctx <<<dim3(BH, 16), 256, 0, stream>>>(K, V, proj, mask, gmax, ksum, ctx);
    perf_tctx<<<(BH * DD * MM) / 256, 256, 0, stream>>>(ctx, ctxT);
    perf_out <<<BH * (NN / 16), 256, 0, stream>>>(Q, proj, ksum, ctxT, out);
}
